// geomGAdisT_85504208929084
// MI455X (gfx1250) — matrix operand order repaired and checked by reading, not compiled, not run on hardware
//
#include <hip/hip_runtime.h>
#include <hip/hip_bf16.h>

// ---------------------------------------------------------------------------
// Problem constants (from reference)
// ---------------------------------------------------------------------------
#define NN      20000            // nodes
#define NE      320000           // edges
#define NEP     (NE + NN)        // edges + self loops
#define NG      256              // graphs
#define DD      64               // feature dim
#define NEG_SLOPE 0.2f

typedef __attribute__((ext_vector_type(16))) __bf16 v16bf;
typedef __attribute__((ext_vector_type(8)))  float  v8f;

// ---------------------------------------------------------------------------
// CDNA5 async global->LDS staging helpers (ASYNCcnt path, inline asm; see
// CDNA5 ISA).  LDS operand is the wave-relative byte
// offset (low 32 bits of the flat shared-pointer).
// ---------------------------------------------------------------------------
__device__ __forceinline__ void async_g2l_b128(const void* gsrc, const void* lds) {
  unsigned l = (unsigned)(unsigned long long)lds;
  asm volatile("global_load_async_to_lds_b128 %0, %1, off"
               :: "v"(l), "v"(gsrc) : "memory");
}
__device__ __forceinline__ void async_g2l_b64(const void* gsrc, const void* lds) {
  unsigned l = (unsigned)(unsigned long long)lds;
  asm volatile("global_load_async_to_lds_b64 %0, %1, off"
               :: "v"(l), "v"(gsrc) : "memory");
}
__device__ __forceinline__ void wait_async0() {
  asm volatile("s_wait_asynccnt 0x0" ::: "memory");
}

// ---------------------------------------------------------------------------
// WMMA helper: accumulate a 16x16 f32 tile:  acc += A(16 x 32*ksteps) * B
// A lives in LDS as bf16 (row-major, leading dim lda).
// B lives in global memory as bf16 (row-major K x N, leading dim ldb).
// Per CDNA5 16-bit layouts:
//   A: lanes 0-15 hold row M=lane: elems 0..7 -> K k0+0..7,  elems 8..15 -> K k0+16..23
//      lanes 16-31 hold row M=lane-16: elems 0..7 -> K k0+8..15, 8..15 -> K k0+24..31
//   B: lane l holds col N=ncol+(l&15), K order same as A
// ---------------------------------------------------------------------------
__device__ __forceinline__ v8f wmma_tile_bf16(const __bf16* __restrict__ Alds, int lda,
                                              const __bf16* __restrict__ Bg, int ldb,
                                              int ncol, int ksteps, v8f acc) {
  const int lane = threadIdx.x & 31;
  const int half = lane >> 4;
  const int m    = lane & 15;
  for (int ks = 0; ks < ksteps; ++ks) {
    const int k0 = ks * 32;
    v16bf a, b;
#pragma unroll
    for (int j = 0; j < 8; ++j) {
      a[j]     = Alds[m * lda + k0 + half * 8 + j];
      a[j + 8] = Alds[m * lda + k0 + 16 + half * 8 + j];
    }
    const __bf16* bcol = Bg + ncol + m;
#pragma unroll
    for (int e = 0; e < 8; ++e) {
      b[e]     = bcol[(size_t)(k0 + half * 8 + e) * ldb];
      b[e + 8] = bcol[(size_t)(k0 + 16 + half * 8 + e) * ldb];
    }
    acc = __builtin_amdgcn_wmma_f32_16x16x32_bf16(false, a, false, b,
                                                  (short)0, acc, false, false);
  }
  return acc;
}

// ---------------------------------------------------------------------------
// Utility kernels
// ---------------------------------------------------------------------------
__global__ void k_fill(float* __restrict__ p, float v, int n) {
  int i = blockIdx.x * blockDim.x + threadIdx.x;
  if (i < n) p[i] = v;
}

__global__ void k_cvt_bf16(const float* __restrict__ src, __bf16* __restrict__ dst, int n) {
  int i = blockIdx.x * blockDim.x + threadIdx.x;
  if (i < n) dst[i] = (__bf16)src[i];
}

// float atomic max via signed/unsigned integer ordering trick
__device__ __forceinline__ void atomicMaxF(float* addr, float v) {
  if (v >= 0.0f) atomicMax((int*)addr, __float_as_int(v));
  else           atomicMin((unsigned int*)addr, __float_as_uint(v));
}

// ---------------------------------------------------------------------------
// h = X @ W   (XB: [NN,64] bf16, W: [64,64] bf16 row-major, h: [NN,64] f32)
// block = 128 threads (4 waves); one 16-row tile per block; wave w -> cols 16w
// The 2KB contiguous A-tile is staged into LDS via the async engine.
// ---------------------------------------------------------------------------
__global__ void k_gemm_node(const __bf16* __restrict__ XB, const __bf16* __restrict__ Wb,
                            float* __restrict__ H) {
  __shared__ __align__(16) __bf16 As[16 * DD];
  const int mt  = blockIdx.x;
  const int tid = threadIdx.x;

  // 16 rows x 64 bf16 = 2048B contiguous; 128 lanes x 16B, one async op each
  const char* g = (const char*)(XB + (size_t)mt * 16 * DD) + tid * 16;
  async_g2l_b128(g, (const char*)As + tid * 16);
  __builtin_prefetch(XB + (size_t)(mt + 1) * 16 * DD, 0, 1);
  wait_async0();
  __syncthreads();

  const int wave = tid >> 5;
  const int ncol = wave * 16;
  v8f acc = {};
  acc = wmma_tile_bf16(As, DD, Wb, DD, ncol, 2, acc);

  const int lane = tid & 31;
  const int half = lane >> 4;
  const int col  = ncol + (lane & 15);
#pragma unroll
  for (int r = 0; r < 8; ++r)
    H[(size_t)(mt * 16 + r + half * 8) * DD + col] = acc[r];
}

// ---------------------------------------------------------------------------
// a_src[n,h] = sum_c h[n,h,c]*att_src[h,c]   (likewise a_dst)
// ---------------------------------------------------------------------------
__global__ void k_att(const float* __restrict__ H,
                      const float* __restrict__ att_s, const float* __restrict__ att_d,
                      float* __restrict__ a_s, float* __restrict__ a_d,
                      int Hh, int Cc) {
  int idx = blockIdx.x * blockDim.x + threadIdx.x;
  if (idx >= NN * Hh) return;
  int n = idx / Hh, hd = idx % Hh;
  const float* hp = H + (size_t)n * DD + hd * Cc;
  const float* sp = att_s + hd * Cc;
  const float* dp = att_d + hd * Cc;
  float ss = 0.f, sd = 0.f;
  for (int c = 0; c < Cc; ++c) { float hv = hp[c]; ss += hv * sp[c]; sd += hv * dp[c]; }
  a_s[idx] = ss;
  a_d[idx] = sd;
}

// ---------------------------------------------------------------------------
// logits + segment max   (edges e<NE from ei, else self loop e-NE)
// ---------------------------------------------------------------------------
__global__ void k_logits(const float* __restrict__ a_s, const float* __restrict__ a_d,
                         const int* __restrict__ ei, float* __restrict__ logits,
                         float* __restrict__ m, int Hh) {
  int idx = blockIdx.x * blockDim.x + threadIdx.x;
  if (idx >= NEP * Hh) return;
  int e = idx / Hh, hd = idx % Hh;
  int s, d;
  if (e < NE) { s = ei[e]; d = ei[NE + e]; } else { s = d = e - NE; }
  float v = a_s[s * Hh + hd] + a_d[d * Hh + hd];
  v = v > 0.f ? v : NEG_SLOPE * v;
  logits[idx] = v;
  atomicMaxF(&m[d * Hh + hd], v);
}

__global__ void k_expsum(const int* __restrict__ ei, float* __restrict__ logits,
                         const float* __restrict__ m, float* __restrict__ den, int Hh) {
  int idx = blockIdx.x * blockDim.x + threadIdx.x;
  if (idx >= NEP * Hh) return;
  int e = idx / Hh, hd = idx % Hh;
  int d = (e < NE) ? ei[NE + e] : (e - NE);
  float p = __expf(logits[idx] - m[d * Hh + hd]);
  logits[idx] = p;
  atomicAdd(&den[d * Hh + hd], p);
}

// out[dst, c] += (p/denom) * h[src, c]   (c in [0,64), head = c / Cc)
__global__ void k_agg(const int* __restrict__ ei, const float* __restrict__ p,
                      const float* __restrict__ den, const float* __restrict__ H,
                      float* __restrict__ out, int Hh, int Cc) {
  int idx = blockIdx.x * blockDim.x + threadIdx.x;
  if (idx >= NEP * DD) return;
  int e = idx >> 6, c = idx & 63;
  int s, d;
  if (e < NE) { s = ei[e]; d = ei[NE + e]; } else { s = d = e - NE; }
  int hd = c / Cc;
  float alpha = p[e * Hh + hd] / den[d * Hh + hd];
  atomicAdd(&out[(size_t)d * DD + c], alpha * H[(size_t)s * DD + c]);
}

__global__ void k_bias_act(float* __restrict__ out, const float* __restrict__ bias, int do_relu) {
  int idx = blockIdx.x * blockDim.x + threadIdx.x;
  if (idx >= NN * DD) return;
  float v = out[idx] + bias[idx & 63];
  if (do_relu) v = v > 0.f ? v : 0.f;
  out[idx] = v;
}

// ---------------------------------------------------------------------------
// Fused edge MLP:  EA = relu([x[row],x[col],EA] @ We1 + be1) @ We2 + be2
// block = 256 threads (8 waves), 16 edges per block, two WMMA GEMMs in LDS.
// A-tile (16x192 bf16) gathered from bf16 side buffers via async engine:
// 48 segments of 64 bf16 (128B) = 16 lanes x 8B each; 3 full-EXEC rounds.
// ---------------------------------------------------------------------------
__global__ void k_edge_mlp(const __bf16* __restrict__ XB, const __bf16* __restrict__ EAB,
                           float* __restrict__ EA, const int* __restrict__ ei,
                           const __bf16* __restrict__ We1b, const float* __restrict__ be1,
                           const __bf16* __restrict__ We2b, const float* __restrict__ be2) {
  __shared__ __align__(16) __bf16 A[16 * 192];
  __shared__ __align__(16) __bf16 H1[16 * 128];
  const int et  = blockIdx.x;
  const int tid = threadIdx.x;

#pragma unroll
  for (int it = 0; it < 3; ++it) {
    int j = it * 256 + tid;          // 0..767
    int q = j >> 4, p = j & 15;      // segment id, 8B slot in segment
    int r = q / 3, seg = q % 3;      // edge row in tile, which 64-wide chunk
    int e = et * 16 + r;
    const char* g;
    if (seg == 0)      g = (const char*)(XB  + (size_t)ei[e]      * DD) + p * 8;
    else if (seg == 1) g = (const char*)(XB  + (size_t)ei[NE + e] * DD) + p * 8;
    else               g = (const char*)(EAB + (size_t)e          * DD) + p * 8;
    async_g2l_b64(g, (const char*)A + (r * 192 + seg * 64) * 2 + p * 8);
  }
  wait_async0();
  __syncthreads();

  const int wave = tid >> 5, lane = tid & 31, half = lane >> 4;

  { // layer 1: [16,192] x [192,128], wave w -> cols 16w, 6 K-steps
    const int ncol = wave * 16;
    v8f acc = {};
    acc = wmma_tile_bf16(A, 192, We1b, 128, ncol, 6, acc);
    const int col = ncol + (lane & 15);
#pragma unroll
    for (int r = 0; r < 8; ++r) {
      float v = acc[r] + be1[col];
      v = v > 0.f ? v : 0.f;
      H1[(r + half * 8) * 128 + col] = (__bf16)v;
    }
  }
  __syncthreads();

  if (wave < 4) { // layer 2: [16,128] x [128,64], waves 0..3, 4 K-steps
    const int ncol = wave * 16;
    v8f acc = {};
    acc = wmma_tile_bf16(H1, 128, We2b, 64, ncol, 4, acc);
    const int col = ncol + (lane & 15);
#pragma unroll
    for (int r = 0; r < 8; ++r)
      EA[(size_t)(et * 16 + r + half * 8) * DD + col] = acc[r] + be2[col];
  }
}

// ---------------------------------------------------------------------------
// Global mean pool + readout MLP
// ---------------------------------------------------------------------------
__global__ void k_pool(const float* __restrict__ X, const int* __restrict__ batch,
                       float* __restrict__ pooled, float* __restrict__ cnt) {
  int idx = blockIdx.x * blockDim.x + threadIdx.x;
  if (idx >= NN * DD) return;
  int n = idx >> 6, c = idx & 63;
  int b = batch[n];
  atomicAdd(&pooled[b * DD + c], X[idx]);
  if (c == 0) atomicAdd(&cnt[b], 1.0f);
}

__global__ void k_readout(const float* __restrict__ pooled, const float* __restrict__ cnt,
                          const float* __restrict__ Wm1, const float* __restrict__ bm1,
                          const float* __restrict__ Wm2, const float* __restrict__ bm2,
                          float* __restrict__ out) {
  int g = blockIdx.x * blockDim.x + threadIdx.x;
  if (g >= NG) return;
  float cn = cnt[g]; cn = cn > 1.0f ? cn : 1.0f;
  float gv[DD];
  for (int i = 0; i < DD; ++i) gv[i] = pooled[g * DD + i] / cn;
  float acc = bm2[0];
  for (int t = 0; t < 32; ++t) {
    float hs = bm1[t];
    for (int i = 0; i < DD; ++i) hs += gv[i] * Wm1[i * 32 + t];
    hs = hs > 0.f ? hs : 0.f;
    acc += hs * Wm2[t];
  }
  out[g] = acc;
}

// ---------------------------------------------------------------------------
// Host orchestration
// ---------------------------------------------------------------------------
static inline int ceil_div(int a, int b) { return (a + b - 1) / b; }

struct GatScratch {
  float *h, *as, *ad, *m, *den, *log;
  __bf16* xb;
};

static void run_gat(const float* xin, const int* ei, const __bf16* Wb,
                    const float* att_s, const float* att_d, const float* bias,
                    int Hh, int Cc, int relu, float* xout,
                    const GatScratch& S, hipStream_t st) {
  k_cvt_bf16<<<ceil_div(NN * DD, 256), 256, 0, st>>>(xin, S.xb, NN * DD);
  k_gemm_node<<<NN / 16, 128, 0, st>>>(S.xb, Wb, S.h);
  k_att<<<ceil_div(NN * Hh, 256), 256, 0, st>>>(S.h, att_s, att_d, S.as, S.ad, Hh, Cc);
  k_fill<<<ceil_div(NN * Hh, 256), 256, 0, st>>>(S.m, -INFINITY, NN * Hh);
  k_fill<<<ceil_div(NN * Hh, 256), 256, 0, st>>>(S.den, 0.0f, NN * Hh);
  k_fill<<<ceil_div(NN * DD, 256), 256, 0, st>>>(xout, 0.0f, NN * DD);
  k_logits<<<ceil_div(NEP * Hh, 256), 256, 0, st>>>(S.as, S.ad, ei, S.log, S.m, Hh);
  k_expsum<<<ceil_div(NEP * Hh, 256), 256, 0, st>>>(ei, S.log, S.m, S.den, Hh);
  k_agg<<<ceil_div(NEP * DD, 256), 256, 0, st>>>(ei, S.log, S.den, S.h, xout, Hh, Cc);
  k_bias_act<<<ceil_div(NN * DD, 256), 256, 0, st>>>(xout, bias, relu);
}

extern "C" void kernel_launch(void* const* d_in, const int* in_sizes, int n_in,
                              void* d_out, int out_size, void* d_ws, size_t ws_size,
                              hipStream_t stream) {
  const float* x0   = (const float*)d_in[0];
  const int*   eix  = (const int*)d_in[1];
  const int*   batch= (const int*)d_in[3];
  const float* ea0  = (const float*)d_in[4];
  const int*   dix  = (const int*)d_in[6];
  const float* W1   = (const float*)d_in[7];
  const float* as1  = (const float*)d_in[8];
  const float* ad1  = (const float*)d_in[9];
  const float* b1   = (const float*)d_in[10];
  const float* W2   = (const float*)d_in[11];
  const float* as2  = (const float*)d_in[12];
  const float* ad2  = (const float*)d_in[13];
  const float* b2   = (const float*)d_in[14];
  const float* We1  = (const float*)d_in[15];
  const float* be1  = (const float*)d_in[16];
  const float* We2  = (const float*)d_in[17];
  const float* be2  = (const float*)d_in[18];
  const float* Wm1  = (const float*)d_in[19];
  const float* bm1  = (const float*)d_in[20];
  const float* Wm2  = (const float*)d_in[21];
  const float* bm2  = (const float*)d_in[22];
  float* out = (float*)d_out;

  // ---- workspace layout (all sizes 256B aligned naturally) ----
  char* w = (char*)d_ws;
  size_t o = 0;
  const size_t SZ_X   = (size_t)NN * DD * sizeof(float);     // 5,120,000
  const size_t SZ_NH  = (size_t)NN * 8 * sizeof(float);      //   640,000
  const size_t SZ_LG  = (size_t)NEP * 8 * sizeof(float);     // 10,880,000
  const size_t SZ_EA  = (size_t)NE * DD * sizeof(float);     // 81,920,000
  const size_t SZ_XB  = (size_t)NN * DD * sizeof(__bf16);    //  2,560,000
  const size_t SZ_EAB = (size_t)NE * DD * sizeof(__bf16);    // 40,960,000

  float* P0 = (float*)(w + o); o += SZ_X;
  float* P1 = (float*)(w + o); o += SZ_X;
  float* P2 = (float*)(w + o); o += SZ_X;
  float* HB = (float*)(w + o); o += SZ_X;
  float* AS = (float*)(w + o); o += SZ_NH;
  float* AD = (float*)(w + o); o += SZ_NH;
  float* MM = (float*)(w + o); o += SZ_NH;
  float* DN = (float*)(w + o); o += SZ_NH;
  float* LG = (float*)(w + o); o += SZ_LG;
  float* EA = (float*)(w + o); o += SZ_EA;
  __bf16* XB  = (__bf16*)(w + o); o += SZ_XB;
  __bf16* EAB = (__bf16*)(w + o); o += SZ_EAB;
  __bf16* W1b  = (__bf16*)(w + o); o += 64 * 64 * sizeof(__bf16);
  __bf16* W2b  = (__bf16*)(w + o); o += 64 * 64 * sizeof(__bf16);
  __bf16* We1b = (__bf16*)(w + o); o += 192 * 128 * sizeof(__bf16);
  __bf16* We2b = (__bf16*)(w + o); o += 128 * 64 * sizeof(__bf16);
  float* POOL = (float*)(w + o); o += NG * DD * sizeof(float);
  float* CNT  = (float*)(w + o); o += 256 * sizeof(float);

  // ---- init: copy inputs we mutate, convert weights to bf16 ----
  hipMemcpyAsync(P0, x0, SZ_X, hipMemcpyDeviceToDevice, stream);
  hipMemcpyAsync(EA, ea0, SZ_EA, hipMemcpyDeviceToDevice, stream);
  k_cvt_bf16<<<ceil_div(64 * 64, 256), 256, 0, stream>>>(W1, W1b, 64 * 64);
  k_cvt_bf16<<<ceil_div(64 * 64, 256), 256, 0, stream>>>(W2, W2b, 64 * 64);
  k_cvt_bf16<<<ceil_div(192 * 128, 256), 256, 0, stream>>>(We1, We1b, 192 * 128);
  k_cvt_bf16<<<ceil_div(128 * 64, 256), 256, 0, stream>>>(We2, We2b, 128 * 64);

  GatScratch S{HB, AS, AD, MM, DN, LG, XB};
  float* xb[3] = {P0, P1, P2};
  int cur = 0;

  for (int layer = 0; layer < 3; ++layer) {
    int a = (cur + 1) % 3, b = (cur + 2) % 3;
    run_gat(xb[cur], dix, W1b, as1, ad1, b1, 8, 8, 1, xb[a], S, stream);
    run_gat(xb[a],   dix, W2b, as2, ad2, b2, 1, 64, 0, xb[b], S, stream);
    run_gat(xb[b],   eix, W1b, as1, ad1, b1, 8, 8, 1, xb[a], S, stream);
    run_gat(xb[a],   eix, W2b, as2, ad2, b2, 1, 64, 0, xb[b], S, stream);
    // edge MLP reads layer-input x (xb[cur]) and updates EA in place
    k_cvt_bf16<<<ceil_div(NN * DD, 256), 256, 0, stream>>>(xb[cur], XB, NN * DD);
    k_cvt_bf16<<<ceil_div(NE * DD, 256), 256, 0, stream>>>(EA, EAB, NE * DD);
    k_edge_mlp<<<NE / 16, 256, 0, stream>>>(XB, EAB, EA, eix, We1b, be1, We2b, be2);
    cur = b;
  }

  // ---- global mean pool + readout ----
  k_fill<<<ceil_div(NG * DD, 256), 256, 0, stream>>>(POOL, 0.0f, NG * DD);
  k_fill<<<1, 256, 0, stream>>>(CNT, 0.0f, NG);
  k_pool<<<ceil_div(NN * DD, 256), 256, 0, stream>>>(xb[cur], batch, POOL, CNT);
  k_readout<<<1, 256, 0, stream>>>(POOL, CNT, Wm1, bm1, Wm2, bm2, out);
}
